// Mtcnn_44470091383485
// MI455X (gfx1250) — matrix operand order repaired and checked by reading, not compiled, not run on hardware
//
#include <hip/hip_runtime.h>
#include <hip/hip_bf16.h>
#include <stdint.h>

#define NBOX 8192
#define NW   256              // 32-bit words per suppression row (8192/32)
#define FLEN 10

typedef _Float16 v16h __attribute__((ext_vector_type(16)));
typedef float    v8f  __attribute__((ext_vector_type(8)));

// ---------------------------------------------------------------------------
// Kernel 1: bitonic sort of (score, idx) descending (ties: larger idx first,
// matching argsort-ascending-stable reversed), then gather sorted box coords.
// ---------------------------------------------------------------------------
__global__ __launch_bounds__(1024) void nms_sort_kernel(
    const float* __restrict__ bboxes, const float* __restrict__ scores,
    int* __restrict__ order,
    float* __restrict__ xs1, float* __restrict__ ys1,
    float* __restrict__ xs2, float* __restrict__ ys2,
    float* __restrict__ ar)
{
    __shared__ float s[NBOX];
    __shared__ int   si[NBOX];
    const int tid = threadIdx.x;

    for (int i = tid; i < NBOX; i += 1024) { s[i] = scores[i]; si[i] = i; }
    __syncthreads();

    for (int k = 2; k <= NBOX; k <<= 1) {
        for (int j = k >> 1; j > 0; j >>= 1) {
            for (int i = tid; i < NBOX; i += 1024) {
                int ixj = i ^ j;
                if (ixj > i) {
                    float a = s[i], b = s[ixj];
                    int   ia = si[i], ib = si[ixj];
                    bool less = (a < b) || (a == b && ia < ib);
                    bool up   = ((i & k) == 0);          // descending segment
                    if (less == up) {
                        s[i] = b; s[ixj] = a;
                        si[i] = ib; si[ixj] = ia;
                    }
                }
            }
            __syncthreads();
        }
    }

    for (int i = tid; i < NBOX; i += 1024) {
        int o = si[i];
        order[i] = o;
        float x = bboxes[o*4+0], y = bboxes[o*4+1];
        float w = bboxes[o*4+2], h = bboxes[o*4+3];
        xs1[i] = x; ys1[i] = y; xs2[i] = x + w; ys2[i] = y + h; ar[i] = w * h;
    }
}

// ---------------------------------------------------------------------------
// Kernel 2: suppression bitmask. One thread = one (i,j) IoU test; wave32
// ballot packs 32 bits into one word. sup[i][w] bit b set iff
// IoU(i, 32w+b) > 0.5 and (32w+b) > i (sorted order).
// grid = (32, 8192), block = 256.
// ---------------------------------------------------------------------------
__global__ __launch_bounds__(256) void nms_sup_kernel(
    const float* __restrict__ xs1, const float* __restrict__ ys1,
    const float* __restrict__ xs2, const float* __restrict__ ys2,
    const float* __restrict__ ar, uint32_t* __restrict__ sup)
{
    const int i = blockIdx.y;
    const int j = blockIdx.x * 256 + threadIdx.x;

    float ix1 = xs1[i], iy1 = ys1[i], ix2 = xs2[i], iy2 = ys2[i], ia = ar[i];
    float jx1 = xs1[j], jy1 = ys1[j], jx2 = xs2[j], jy2 = ys2[j], ja = ar[j];

    float w = fminf(ix2, jx2) - fmaxf(ix1, jx1);
    float h = fminf(iy2, jy2) - fmaxf(iy1, jy1);
    float inter = fmaxf(w, 0.0f) * fmaxf(h, 0.0f);
    // inter/(ia+ja-inter) > 0.5  <=>  inter > 0.5*(ia+ja-inter)
    bool bit = (inter > 0.5f * (ia + ja - inter)) && (j > i);

    unsigned m = (unsigned)__builtin_amdgcn_ballot_w32(bit);
    if ((threadIdx.x & 31) == 0)
        sup[(size_t)i * NW + (j >> 5)] = m;
}

// ---------------------------------------------------------------------------
// Kernel 3: sequential greedy scan, single wave32. removed mask lives in LDS
// (256 words, 8 per lane). Since row i only sets bits j>i, the final mask
// equals the at-visit mask: keep = ~removed. Same-wave LDS ops are in-order,
// so no barriers inside the loop.
// ---------------------------------------------------------------------------
__global__ __launch_bounds__(32) void nms_scan_kernel(
    const uint32_t* __restrict__ sup, const int* __restrict__ order,
    float* __restrict__ keepf)
{
    volatile __shared__ uint32_t removed[NW];
    const int lane = threadIdx.x;

    for (int w = lane; w < NW; w += 32) removed[w] = 0u;
    __syncthreads();   // single-wave WG: lowers to NOP, LDS already in-order

    volatile uint32_t* dst = &removed[lane * 8];

    for (int i = 0; i < NBOX; ++i) {
        const uint32_t* row = sup + (size_t)i * NW + lane * 8;
        // issue row loads early (independent of the alive test)
        uint4 r0 = *(const uint4*)(row);
        uint4 r1 = *(const uint4*)(row + 4);
        if (i + 1 < NBOX)
            __builtin_prefetch(row + NW, 0, 3);   // global_prefetch_b8, next row

        uint32_t wrd = removed[i >> 5];           // broadcast LDS read
        bool alive = ((wrd >> (i & 31)) & 1u) == 0u;   // uniform across lanes
        if (alive) {
            dst[0] |= r0.x; dst[1] |= r0.y; dst[2] |= r0.z; dst[3] |= r0.w;
            dst[4] |= r1.x; dst[5] |= r1.y; dst[6] |= r1.z; dst[7] |= r1.w;
        }
    }

    // scatter keep mask back to original box indices as a float multiplier
    for (int i = lane; i < NBOX; i += 32) {
        uint32_t wrd = removed[i >> 5];
        bool rem = (wrd >> (i & 31)) & 1u;
        keepf[order[i]] = rem ? 0.0f : 1.0f;
    }
}

// ---------------------------------------------------------------------------
// Kernel 4: bbox regression via WMMA. One wave per 16-row tile:
// D = A(16x32 f16, K padded 10->32) x B(32x16 f16, cols padded 4->16) + bias.
// A and B share contiguous K = 16*half + e. All fragment loads are unconditional
// (always in-bounds) and masked with selects, so EXEC is full and there are
// no divergent branches before the WMMA. Epilogue: fast v_rcp_f32 sigmoid and
// an arithmetic-masked (multiply-accumulate) column select that cannot be
// turned back into control flow.
// ---------------------------------------------------------------------------
__global__ __launch_bounds__(256) void reg_kernel(
    const float* __restrict__ features, const float* __restrict__ W,
    const float* __restrict__ bias, const float* __restrict__ bboxes,
    const float* __restrict__ keepf, const int* __restrict__ widthp,
    const int* __restrict__ heightp, float* __restrict__ out)
{
    const int tid  = threadIdx.x;
    const int tile = blockIdx.x * 8 + (tid >> 5);
    const int lane = tid & 31;
    const int m0   = tile * 16;
    const int half = lane >> 4;     // 0: lanes 0-15, 1: lanes 16-31
    const int sub  = lane & 15;
    const bool h1  = (half != 0);

    // ---- A fragment: 16-bit A-matrix 16x32, contiguous K order, same as B.
    // Lane holds row M = sub (both halves). Element e maps to K = 16*half + e;
    // only K<10 is non-zero (zero padding). Load the full 10-float row
    // unconditionally (rows are 8B-aligned, stride 40B), mask with selects.
    const int row = m0 + sub;
    const float2* fp = (const float2*)(features + row * FLEN);
    float2 f01 = fp[0], f23 = fp[1], f45 = fp[2], f67 = fp[3], f89 = fp[4];
    float fr[FLEN] = { f01.x, f01.y, f23.x, f23.y, f45.x,
                       f45.y, f67.x, f67.y, f89.x, f89.y };
    v16h a;
#pragma unroll
    for (int e = 0; e < 16; ++e) {
        float v = 0.0f;
        if (e < FLEN) {
            v = h1 ? 0.0f : fr[e];                      // half0: K = e; half1: K = 16 + e (pad)
        }
        a[e] = (_Float16)v;
    }

    // ---- B fragment: 32x16 (KxN), rows striped across lanes (lanes 0-15 hold
    // K=0..15 for column N=lane; lanes 16-31 hold K=16..31). Element e -> K =
    // 16*half + e; only K<10 and N<4 are non-zero. Loads always in-bounds.
    const int  c4   = sub & 3;
    const bool bact = (sub < 4) && !h1;
    float wr[FLEN];
#pragma unroll
    for (int k = 0; k < FLEN; ++k) wr[k] = W[k * 4 + c4];
    v16h b;
#pragma unroll
    for (int e = 0; e < 16; ++e) {
        float v = (e < FLEN) ? (bact ? wr[e] : 0.0f) : 0.0f;
        b[e] = (_Float16)v;
    }

    // ---- C accumulator pre-loaded with bias[n] on valid columns.
    float bload = bias[c4];
    float cb    = (sub < 4) ? bload : 0.0f;
    v8f c;
#pragma unroll
    for (int r = 0; r < 8; ++r) c[r] = cb;

    c = __builtin_amdgcn_wmma_f32_16x16x32_f16(
            /*neg_a=*/false, a, /*neg_b=*/false, b,
            /*c_mod=*/(short)0, c, /*reuse_a=*/false, /*reuse_b=*/false);

    const float width  = (float)widthp[0];
    const float height = (float)heightp[0];

    // 0/1 column masks (lower to v_cndmask of constants)
    const float mx = (sub == 0) ? 1.0f : 0.0f;
    const float my = (sub == 1) ? 1.0f : 0.0f;
    const float mw = (sub == 2) ? 1.0f : 0.0f;
    const float mh = (sub == 3) ? 1.0f : 0.0f;

    if (sub < 4) {   // single divergence point for the whole epilogue
#pragma unroll
        for (int r = 0; r < 8; ++r) {
            int m = m0 + r + 8 * half;   // C/D layout: VGPR r -> M=r (+8 hi half)
            float t  = __builtin_amdgcn_rcpf(1.0f + __expf(-c[r])); // v_rcp_f32
            float et = __expf(t);
            float rx = bboxes[m*4+0], ry = bboxes[m*4+1];
            float rw = bboxes[m*4+2], rh = bboxes[m*4+3];
            float v0 = fmaxf(t * rw + rx, 0.0f) * width;
            float v1 = fmaxf(t * rh + ry, 0.0f) * height;
            float v2 = rw * et * width;
            float v3 = rh * et * height;
            // arithmetic-masked select: stays straight-line FMA code
            float v  = v0 * mx + v1 * my + v2 * mw + v3 * mh;
            out[m * 4 + sub] = v * keepf[m];
        }
    }
}

// ---------------------------------------------------------------------------
// Host-side launcher
// ---------------------------------------------------------------------------
extern "C" void kernel_launch(void* const* d_in, const int* in_sizes, int n_in,
                              void* d_out, int out_size, void* d_ws, size_t ws_size,
                              hipStream_t stream) {
    const float* bboxes   = (const float*)d_in[0];
    const float* scores   = (const float*)d_in[1];
    const float* features = (const float*)d_in[2];
    const float* W        = (const float*)d_in[3];
    const float* bias     = (const float*)d_in[4];
    const int*   widthp   = (const int*)d_in[5];
    const int*   heightp  = (const int*)d_in[6];
    float*       out      = (float*)d_out;

    char* ws = (char*)d_ws;
    // workspace layout (bytes)
    int*      order = (int*)     (ws + 0);
    float*    xs1   = (float*)   (ws + 32768);
    float*    ys1   = (float*)   (ws + 65536);
    float*    xs2   = (float*)   (ws + 98304);
    float*    ys2   = (float*)   (ws + 131072);
    float*    ar    = (float*)   (ws + 163840);
    float*    keepf = (float*)   (ws + 196608);
    uint32_t* sup   = (uint32_t*)(ws + 229376);   // 8192*256*4 = 8 MB

    nms_sort_kernel<<<1, 1024, 0, stream>>>(bboxes, scores, order,
                                            xs1, ys1, xs2, ys2, ar);

    nms_sup_kernel<<<dim3(32, NBOX), 256, 0, stream>>>(xs1, ys1, xs2, ys2, ar, sup);

    nms_scan_kernel<<<1, 32, 0, stream>>>(sup, order, keepf);

    reg_kernel<<<NBOX / 128, 256, 0, stream>>>(features, W, bias, bboxes,
                                               keepf, widthp, heightp, out);
}
